// Attention_9268539425019
// MI455X (gfx1250) — compile-verified, matrix operand order checked by reading, not run on hardware
//
#include <hip/hip_runtime.h>
#include <hip/hip_bf16.h>

// ---------------- problem constants ----------------
#define BB     2
#define SS     2048
#define DIMM   4096
#define NH     32
#define NKV    8
#define HD     128
#define NREP   4          // NH / NKV
#define KVDIM  (NKV*HD)   // 1024

typedef __bf16 bf16_t;
typedef bf16_t v16bf __attribute__((ext_vector_type(16)));
typedef bf16_t v8bf  __attribute__((ext_vector_type(8)));
typedef float  v8f   __attribute__((ext_vector_type(8)));
typedef unsigned int u32x4 __attribute__((ext_vector_type(4)));
typedef unsigned int u32x8 __attribute__((ext_vector_type(8)));

// Build a 16-bit WMMA A/B fragment: 8 contiguous bf16 at p, 8 more at p+16.
// (Per CDNA5 ISA 7.12.2: element j holds K = (j&7) + 16*(j>>3) + 8*(lane>>4);
// caller supplies p already offset by 8*(lane>>4).)
__device__ __forceinline__ v16bf make_frag(const bf16_t* p) {
    v8bf lo = *(const v8bf*)(p);
    v8bf hi = *(const v8bf*)(p + 16);
    return __builtin_shufflevector(lo, hi, 0,1,2,3,4,5,6,7,8,9,10,11,12,13,14,15);
}

// ---------------- Tensor Data Mover: 2-D bf16 tile -> LDS ----------------
// Issues TENSOR_LOAD_TO_LDS with a 2-group D# (VADDR2/3 = NULL, <=2D tensor).
// tile: rows x cols bf16, row stride = rowStride elements, into *lptr (LDS).
// The LDS pointer is passed to the asm (unreferenced) so the compiler knows
// the LDS object is written by the DMA and keeps/orders the ds_load reads.
// All arguments must be wave-uniform. Tracked by TENSORcnt.
__device__ __forceinline__ void tdm_load_tile_2d(bf16_t* lptr,
                                                 const bf16_t* gptr,
                                                 int cols, int rows,
                                                 int rowStride) {
    // Generic pointer to LDS: addr = {SHARED_BASE, lds_offset} -> low 32 bits.
    unsigned lds_off = (unsigned)(uintptr_t)lptr;
    unsigned long long ga = (unsigned long long)(uintptr_t)gptr;
    u32x4 g0;
    g0.x = 1u;                                    // count=1, is_restore=0, gather=0
    g0.y = lds_off;                               // lds_addr (bytes)
    g0.z = (unsigned)ga;                          // global_addr[31:0]
    g0.w = ((unsigned)(ga >> 32) & 0x01FFFFFFu)   // global_addr[56:32]
           | (2u << 30);                          // type=2 ("image")
    u32x8 g1;
    g1[0] = (1u << 16);                           // data_size=1 (2 bytes/elem)
    g1[1] = ((unsigned)rowStride & 0xFFFFu) << 16;          // tensor_dim0[15:0]
    g1[2] = (((unsigned)rowStride >> 16) & 0xFFFFu)         // tensor_dim0[31:16]
           | (((unsigned)rows & 0xFFFFu) << 16);            // tensor_dim1[15:0]
    g1[3] = (((unsigned)rows >> 16) & 0xFFFFu)              // tensor_dim1[31:16]
           | (((unsigned)cols & 0xFFFFu) << 16);            // tile_dim0
    g1[4] = (unsigned)rows & 0xFFFFu;                       // tile_dim1 (tile_dim2=0)
    g1[5] = (unsigned)rowStride;                            // tensor_dim0_stride[31:0]
    g1[6] = 0u;                                             // stride0[47:32], stride1[15:0]
    g1[7] = 0u;                                             // stride1[47:16]
    asm volatile("tensor_load_to_lds %0, %1"
                 :: "s"(g0), "s"(g1), "v"(lptr)
                 : "memory");
}

// ---------------- fp32 -> bf16 convert ----------------
__global__ void f32_to_bf16_kernel(const float* __restrict__ in,
                                   bf16_t* __restrict__ out, int n) {
    int i = blockIdx.x * blockDim.x + threadIdx.x;
    if (i < n) out[i] = (bf16_t)in[i];
}

// ---------------- bf16 WMMA GEMM:  C[M,N] = A[M,K] * B[N,K]^T ----------------
// block = 256 threads (8 waves). Block tile: 128(M) x 64(N). Wave w owns rows
// [blockM + 16w, +16). The 64x64 bf16 B tile (shared by all 8 waves) is
// streamed into LDS by the Tensor Data Mover, double-buffered; A fragments are
// read directly from global (b128 clauses). K stepped by 64 per TDM chunk.
__global__ __launch_bounds__(256)
void gemm_bf16_wmma(const bf16_t* __restrict__ A,
                    const bf16_t* __restrict__ Bw,
                    float* __restrict__ C,
                    int M, int N, int K) {
    __shared__ bf16_t btile[2][64 * 64];          // 2 x 8KB double buffer

    const int lane = threadIdx.x & 31;
    const int w    = threadIdx.x >> 5;
    const int ln   = lane & 15;
    const int hl   = lane >> 4;

    const int rowBase = blockIdx.y * 128 + w * 16;
    const int colBase = blockIdx.x * 64;

    v8f acc[4] = {};

    const bf16_t* ap    = A  + (size_t)(rowBase + ln) * K + 8 * hl;
    const bf16_t* bbase = Bw + (size_t)colBase * K;

    const int nk = K >> 6;                        // 64-wide K chunks

    // prologue: TDM fills buffer 0 (wave 0 drives the DMA for the block)
    if (w == 0) {
        tdm_load_tile_2d(&btile[0][0], bbase, 64, 64, K);
        __builtin_amdgcn_s_wait_tensorcnt(0);
    }
    __syncthreads();

    for (int c = 0; c < nk; ++c) {
        const int cur = c & 1;
        if (w == 0 && (c + 1) < nk)               // async: fill other buffer
            tdm_load_tile_2d(&btile[cur ^ 1][0],
                             bbase + (size_t)(c + 1) * 64, 64, 64, K);
        if ((c + 1) < nk)                         // prefetch next A chunk
            __builtin_prefetch(ap + (size_t)(c + 1) * 64, 0, 3);

        const bf16_t* lb = &btile[cur][0];
#pragma unroll
        for (int ks = 0; ks < 2; ++ks) {
            v16bf af = make_frag(ap + c * 64 + ks * 32);
#pragma unroll
            for (int nt = 0; nt < 4; ++nt) {
                v16bf bfr = make_frag(lb + (nt * 16 + ln) * 64 + ks * 32 + 8 * hl);
                acc[nt] = __builtin_amdgcn_wmma_f32_16x16x32_bf16(
                    false, af, false, bfr, (short)0, acc[nt], false, false);
            }
        }

        if (w == 0) __builtin_amdgcn_s_wait_tensorcnt(0);
        __syncthreads();                          // publish next buffer / retire cur
    }

#pragma unroll
    for (int nt = 0; nt < 4; ++nt)
#pragma unroll
        for (int v = 0; v < 8; ++v)
            C[(size_t)(rowBase + v + 8 * hl) * N + colBase + nt * 16 + ln] = acc[nt][v];
}

// ---------------- RoPE + repack Q: [B*S, DIM] f32 -> [B,H,S,HD] bf16 --------
__global__ void rope_pack_q(const float* __restrict__ Qf,
                            const float* __restrict__ cosT,
                            const float* __restrict__ sinT,
                            bf16_t* __restrict__ Qb) {
    int idx = blockIdx.x * blockDim.x + threadIdx.x;   // B*S*NH*64 threads
    int j = idx & 63; int r = idx >> 6;
    int h = r & 31;   r >>= 5;
    int s = r & 2047; int b = r >> 11;
    if (b >= BB) return;
    size_t src = (size_t)(b * SS + s) * DIMM + h * HD + 2 * j;
    float xr = Qf[src], xi = Qf[src + 1];
    float c  = cosT[s * 64 + j], sn = sinT[s * 64 + j];
    size_t dst = ((size_t)(b * NH + h) * SS + s) * HD + 2 * j;
    Qb[dst]     = (bf16_t)(xr * c - xi * sn);
    Qb[dst + 1] = (bf16_t)(xr * sn + xi * c);
}

// ---------------- RoPE + repack K: [B*S, KVDIM] f32 -> [B,NKV,S,HD] bf16 ----
__global__ void rope_pack_k(const float* __restrict__ Kf,
                            const float* __restrict__ cosT,
                            const float* __restrict__ sinT,
                            bf16_t* __restrict__ Kb) {
    int idx = blockIdx.x * blockDim.x + threadIdx.x;   // B*S*NKV*64 threads
    int j = idx & 63; int r = idx >> 6;
    int h = r & 7;    r >>= 3;
    int s = r & 2047; int b = r >> 11;
    if (b >= BB) return;
    size_t src = (size_t)(b * SS + s) * KVDIM + h * HD + 2 * j;
    float xr = Kf[src], xi = Kf[src + 1];
    float c  = cosT[s * 64 + j], sn = sinT[s * 64 + j];
    size_t dst = ((size_t)(b * NKV + h) * SS + s) * HD + 2 * j;
    Kb[dst]     = (bf16_t)(xr * c - xi * sn);
    Kb[dst + 1] = (bf16_t)(xr * sn + xi * c);
}

// ---------------- pack V transposed: [B*S, KVDIM] f32 -> [B,NKV,HD,S] bf16 --
__global__ void pack_v_t(const float* __restrict__ Vf, bf16_t* __restrict__ Vt) {
    int idx = blockIdx.x * blockDim.x + threadIdx.x;   // B*S*NKV*HD threads
    int d = idx & 127; int r = idx >> 7;
    int h = r & 7;     r >>= 3;
    int s = r & 2047;  int b = r >> 11;
    if (b >= BB) return;
    float v = Vf[(size_t)(b * SS + s) * KVDIM + h * HD + d];
    Vt[((size_t)(b * NKV + h) * HD + d) * SS + s] = (bf16_t)v;
}

// ---------------- flash attention (causal, GQA) ----------------
// block = 256 (8 waves); wave w handles query tile qTile = blockIdx.x*8 + w of
// head blockIdx.y, batch blockIdx.z. Online softmax; P re-layout via LDS.
__global__ __launch_bounds__(256)
void flash_attn(const bf16_t* __restrict__ Qb,   // [B,NH,S,HD]
                const bf16_t* __restrict__ Kb,   // [B,NKV,S,HD]
                const bf16_t* __restrict__ Vt,   // [B,NKV,HD,S]
                bf16_t* __restrict__ Ob) {       // [B*S, DIM]
    __shared__ bf16_t ldsP[8 * 16 * 32];

    const int lane = threadIdx.x & 31;
    const int w    = threadIdx.x >> 5;
    const int ln   = lane & 15;
    const int hl   = lane >> 4;
    const int qTile = blockIdx.x * 8 + w;
    const int h  = blockIdx.y;
    const int b  = blockIdx.z;
    const int kh = h >> 2;              // NREP = 4
    const int q0 = qTile * 16;

    const bf16_t* Qbase = Qb + ((size_t)(b * NH  + h ) * SS) * HD;
    const bf16_t* Kbase = Kb + ((size_t)(b * NKV + kh) * SS) * HD;
    const bf16_t* Vbase = Vt + ((size_t)(b * NKV + kh) * HD) * SS;
    bf16_t* Pl = ldsP + w * (16 * 32);

    // Q fragments across d = 4 chunks of 32
    v16bf qf[4];
    {
        const bf16_t* qp = Qbase + (size_t)(q0 + ln) * HD + 8 * hl;
#pragma unroll
        for (int dc = 0; dc < 4; ++dc) qf[dc] = make_frag(qp + dc * 32);
    }

    float m_i[8], l_i[8];
#pragma unroll
    for (int v = 0; v < 8; ++v) { m_i[v] = -__builtin_inff(); l_i[v] = 0.f; }
    v8f o[8] = {};

    const float scale = 0.08838834764831845f;   // 1/sqrt(128)
    const float L2E   = 1.4426950408889634f;

    for (int kc = 0; kc * 32 <= q0 + 15; ++kc) {
#pragma unroll
        for (int t = 0; t < 2; ++t) {
            const int kk0 = kc * 32 + t * 16;
            const bool live = (kk0 <= q0 + 15);
            v8f sacc = {};
            if (live) {
                const bf16_t* kp = Kbase + (size_t)(kk0 + ln) * HD + 8 * hl;
#pragma unroll
                for (int dc = 0; dc < 4; ++dc) {
                    v16bf kf = make_frag(kp + dc * 32);
                    sacc = __builtin_amdgcn_wmma_f32_16x16x32_bf16(
                        false, qf[dc], false, kf, (short)0, sacc, false, false);
                }
            }
            const int kg = kk0 + ln;
#pragma unroll
            for (int v = 0; v < 8; ++v) {
                const int qg = q0 + v + 8 * hl;
                float s = live ? sacc[v] * scale : -__builtin_inff();
                if (kg > qg) s = -__builtin_inff();
                // row max across the 16 lanes holding this row
                float r = s;
                for (int off = 8; off >= 1; off >>= 1)
                    r = fmaxf(r, __shfl_xor(r, off, 16));
                const float mn    = fmaxf(m_i[v], r);
                const float alpha = exp2f((m_i[v] - mn) * L2E);
                const float p     = exp2f((s - mn) * L2E);
                float ps = p;
                for (int off = 8; off >= 1; off >>= 1)
                    ps += __shfl_xor(ps, off, 16);
                l_i[v] = l_i[v] * alpha + ps;
                m_i[v] = mn;
#pragma unroll
                for (int dt = 0; dt < 8; ++dt) o[dt][v] *= alpha;
                Pl[(v + 8 * hl) * 32 + t * 16 + ln] = (bf16_t)p;
            }
        }
        // reload P as an A-fragment (per-wave LDS region; DS ops in-order)
        v16bf pf = make_frag(Pl + ln * 32 + 8 * hl);
#pragma unroll
        for (int dt = 0; dt < 8; ++dt) {
            const bf16_t* vp = Vbase + (size_t)(dt * 16 + ln) * SS + kc * 32 + 8 * hl;
            v16bf vf = make_frag(vp);
            o[dt] = __builtin_amdgcn_wmma_f32_16x16x32_bf16(
                false, pf, false, vf, (short)0, o[dt], false, false);
        }
    }

#pragma unroll
    for (int dt = 0; dt < 8; ++dt)
#pragma unroll
        for (int v = 0; v < 8; ++v) {
            const int qg = q0 + v + 8 * hl;
            const float val = o[dt][v] / l_i[v];
            Ob[(size_t)(b * SS + qg) * DIMM + h * HD + dt * 16 + ln] = (bf16_t)val;
        }
}

// ---------------- host launcher ----------------
extern "C" void kernel_launch(void* const* d_in, const int* in_sizes, int n_in,
                              void* d_out, int out_size, void* d_ws, size_t ws_size,
                              hipStream_t stream) {
    const float* x  = (const float*)d_in[0];
    const float* wq = (const float*)d_in[1];
    const float* wk = (const float*)d_in[2];
    const float* wv = (const float*)d_in[3];
    const float* wo = (const float*)d_in[4];
    const float* fc = (const float*)d_in[5];
    const float* fs = (const float*)d_in[6];
    float* out = (float*)d_out;

    const size_t MS = (size_t)BB * SS;   // 4096 rows

    uint8_t* w8 = (uint8_t*)d_ws;
    size_t off = 0;
    auto carve = [&](size_t bytes) -> void* {
        void* p = w8 + off;
        off += (bytes + 255) & ~(size_t)255;
        return p;
    };
    bf16_t* xb   = (bf16_t*)carve(MS * DIMM * sizeof(bf16_t));
    bf16_t* wqb  = (bf16_t*)carve((size_t)DIMM * DIMM * sizeof(bf16_t));
    bf16_t* wkb  = (bf16_t*)carve((size_t)KVDIM * DIMM * sizeof(bf16_t));
    bf16_t* wvb  = (bf16_t*)carve((size_t)KVDIM * DIMM * sizeof(bf16_t));
    bf16_t* wob  = (bf16_t*)carve((size_t)DIMM * DIMM * sizeof(bf16_t));
    float*  Qf   = (float*)carve(MS * DIMM * sizeof(float));
    float*  Kf   = (float*)carve(MS * KVDIM * sizeof(float));
    float*  Vf   = (float*)carve(MS * KVDIM * sizeof(float));
    bf16_t* Qbf  = (bf16_t*)carve(MS * DIMM * sizeof(bf16_t));
    bf16_t* Kbf  = (bf16_t*)carve(MS * KVDIM * sizeof(bf16_t));
    bf16_t* Vtb  = (bf16_t*)carve(MS * KVDIM * sizeof(bf16_t));
    bf16_t* attn = (bf16_t*)carve(MS * DIMM * sizeof(bf16_t));
    (void)ws_size; (void)in_sizes; (void)n_in; (void)out_size;

    const int TPB = 256;
    auto cdiv = [](size_t a, int b) { return (int)((a + b - 1) / b); };

    // 1) convert to bf16
    f32_to_bf16_kernel<<<cdiv(MS * DIMM, TPB), TPB, 0, stream>>>(x,  xb,  (int)(MS * DIMM));
    f32_to_bf16_kernel<<<cdiv((size_t)DIMM * DIMM, TPB), TPB, 0, stream>>>(wq, wqb, DIMM * DIMM);
    f32_to_bf16_kernel<<<cdiv((size_t)KVDIM * DIMM, TPB), TPB, 0, stream>>>(wk, wkb, KVDIM * DIMM);
    f32_to_bf16_kernel<<<cdiv((size_t)KVDIM * DIMM, TPB), TPB, 0, stream>>>(wv, wvb, KVDIM * DIMM);
    f32_to_bf16_kernel<<<cdiv((size_t)DIMM * DIMM, TPB), TPB, 0, stream>>>(wo, wob, DIMM * DIMM);

    // 2) QKV projections (C = A * W^T), TDM-staged B tiles
    gemm_bf16_wmma<<<dim3(DIMM / 64,  (int)(MS / 128)), TPB, 0, stream>>>(xb, wqb, Qf, (int)MS, DIMM,  DIMM);
    gemm_bf16_wmma<<<dim3(KVDIM / 64, (int)(MS / 128)), TPB, 0, stream>>>(xb, wkb, Kf, (int)MS, KVDIM, DIMM);
    gemm_bf16_wmma<<<dim3(KVDIM / 64, (int)(MS / 128)), TPB, 0, stream>>>(xb, wvb, Vf, (int)MS, KVDIM, DIMM);

    // 3) RoPE + repack
    rope_pack_q<<<cdiv((size_t)BB * SS * NH  * 64, TPB), TPB, 0, stream>>>(Qf, fc, fs, Qbf);
    rope_pack_k<<<cdiv((size_t)BB * SS * NKV * 64, TPB), TPB, 0, stream>>>(Kf, fc, fs, Kbf);
    pack_v_t   <<<cdiv((size_t)BB * SS * NKV * HD, TPB), TPB, 0, stream>>>(Vf, Vtb);

    // 4) flash attention: 128 q-tiles / 8 waves per block = 16 blocks in x
    flash_attn<<<dim3(SS / 16 / 8, NH, BB), TPB, 0, stream>>>(Qbf, Kbf, Vtb, attn);

    // 5) output projection
    gemm_bf16_wmma<<<dim3(DIMM / 64, (int)(MS / 128)), TPB, 0, stream>>>(attn, wob, out, (int)MS, DIMM, DIMM);
}
